// ScaffoldPointLoRA_46024869544429
// MI455X (gfx1250) — matrix operand order repaired and checked by reading, not compiled, not run on hardware
//
#include <hip/hip_runtime.h>
#include <math.h>

typedef __bf16 bf16_t;
typedef bf16_t v16bf __attribute__((ext_vector_type(16)));
typedef bf16_t v8bf  __attribute__((ext_vector_type(8)));
typedef float  v8f   __attribute__((ext_vector_type(8)));

#define NPTS  8192
#define DFEAT 768
#define NB    2

__device__ __forceinline__ bf16_t f2bf(float f) {
  union { float f; unsigned u; } v; v.f = f;
  unsigned r = v.u + 0x7FFFu + ((v.u >> 16) & 1u);   // round-to-nearest-even
  unsigned short h = (unsigned short)(r >> 16);
  return __builtin_bit_cast(bf16_t, h);
}

// ---------------------------------------------------------------------------
// Farthest point sampling: one workgroup per batch, min-distance array in LDS.
// fps(coords,128)/fps(coords,64) are prefixes of fps(coords,256).
// ---------------------------------------------------------------------------
__global__ void fps_kernel(const float* __restrict__ coords, int* __restrict__ fps_idx) {
  __shared__ float smind[NPTS];
  __shared__ float sval[256];
  __shared__ int   sidx[256];
  __shared__ int   snxt;
  const int b = blockIdx.x, t = threadIdx.x;
  const float* C = coords + (size_t)b * NPTS * 3;
  for (int p = t; p < NPTS; p += 256) smind[p] = 3.4028235e38f;
  if (t == 0) fps_idx[b * 256] = 0;
  __syncthreads();
  int last = 0;
  for (int i = 1; i < 256; ++i) {
    const float lx = C[last*3+0], ly = C[last*3+1], lz = C[last*3+2];
    float best = -1.0f; int bi = NPTS;
    for (int p = t; p < NPTS; p += 256) {
      float dx = C[p*3+0]-lx, dy = C[p*3+1]-ly, dz = C[p*3+2]-lz;
      float d  = sqrtf(dx*dx + dy*dy + dz*dz);
      float m  = fminf(smind[p], d);
      smind[p] = m;
      if (m > best || (m == best && p < bi)) { best = m; bi = p; }
    }
    sval[t] = best; sidx[t] = bi;
    __syncthreads();
    for (int s = 128; s > 0; s >>= 1) {
      if (t < s) {
        float v2 = sval[t+s]; int i2 = sidx[t+s];
        if (v2 > sval[t] || (v2 == sval[t] && i2 < sidx[t])) { sval[t] = v2; sidx[t] = i2; }
      }
      __syncthreads();
    }
    if (t == 0) { int nx = sidx[0]; snxt = nx; smind[nx] = 0.0f; fps_idx[b*256 + i] = nx; }
    __syncthreads();
    last = snxt;
  }
}

// z mean + var(ddof=1) per batch
__global__ void zstats_kernel(const float* __restrict__ coords, float* __restrict__ zstat) {
  __shared__ float ss[256], sq[256];
  const int b = blockIdx.x, t = threadIdx.x;
  const float* C = coords + (size_t)b * NPTS * 3;
  float s = 0.f, q = 0.f;
  for (int p = t; p < NPTS; p += 256) { float z = C[p*3+2]; s += z; q += z*z; }
  ss[t] = s; sq[t] = q; __syncthreads();
  for (int h = 128; h > 0; h >>= 1) { if (t < h) { ss[t] += ss[t+h]; sq[t] += sq[t+h]; } __syncthreads(); }
  if (t == 0) {
    float mean = ss[0] / (float)NPTS;
    float var  = (sq[0] - ss[0]*ss[0] / (float)NPTS) / (float)(NPTS - 1);
    zstat[b*2+0] = mean; zstat[b*2+1] = var;
  }
}

// density only at the 64 detail centers (64 x 8192 instead of 8192^2)
__global__ void density_kernel(const float* __restrict__ coords, const int* __restrict__ fps_idx,
                               float* __restrict__ density) {
  __shared__ int scnt[256];
  const int c = blockIdx.x, b = blockIdx.y, t = threadIdx.x;
  const float* C = coords + (size_t)b * NPTS * 3;
  const int pid = fps_idx[b*256 + c];
  const float cx = C[pid*3+0], cy = C[pid*3+1], cz = C[pid*3+2];
  int cnt = 0;
  for (int p = t; p < NPTS; p += 256) {
    float dx = C[p*3+0]-cx, dy = C[p*3+1]-cy, dz = C[p*3+2]-cz;
    cnt += (dx*dx + dy*dy + dz*dz < 0.25f) ? 1 : 0;
  }
  scnt[t] = cnt; __syncthreads();
  for (int h = 128; h > 0; h >>= 1) { if (t < h) scnt[t] += scnt[t+h]; __syncthreads(); }
  if (t == 0) density[b*64 + c] = (float)scnt[0];
}

__global__ void safety_kernel(const float* __restrict__ coords, const int* __restrict__ fps_idx,
                              const float* __restrict__ zstat, const float* __restrict__ density,
                              float* __restrict__ sg, float* __restrict__ sc, float* __restrict__ sd) {
  const int b = blockIdx.x, t = threadIdx.x;
  const float mean = zstat[b*2+0], var = zstat[b*2+1];
  const int pid = fps_idx[b*256 + t];
  const float z = coords[((size_t)b*NPTS + pid)*3 + 2];
  sg[b*256 + t] = 1.0f + 0.95f / (1.0f + expf(-(z - mean) * 0.2f));   // P_JOINT
  sc[b*256 + t] = 1.0f + 0.9f  * expf(-var * 10.0f);                  // P_PLATFORM
  if (t < 64) sd[b*64 + t] = 1.0f + density[b*64 + t] * (0.95f / (float)NPTS);
}

__global__ void cvt_f32_bf16(const float* __restrict__ s, bf16_t* __restrict__ d, int n) {
  int i = blockIdx.x * blockDim.x + threadIdx.x;
  if (i < n) d[i] = f2bf(s[i]);
}

__global__ void gather_cfeat_kernel(const float* __restrict__ feat, const int* __restrict__ fps_idx,
                                    bf16_t* __restrict__ cfeat) {
  const int r = blockIdx.x, b = blockIdx.y, t = threadIdx.x;
  const int pid = fps_idx[b*256 + r];
  const float* src = feat + ((size_t)b*NPTS + pid) * DFEAT;
  bf16_t* dst = cfeat + ((size_t)b*256 + r) * DFEAT;
  for (int d = t; d < DFEAT; d += 256) dst[d] = f2bf(src[d]);
}

// ---------------------------------------------------------------------------
// Generic bf16 WMMA GEMM: out[b] = act(A[b] (MxK, K-contig) @ W^T (NxK, K-contig) + bias)
// One wave per 16x16 tile; K loop in steps of 32 on v_wmma_f32_16x16x32_bf16.
// Fragment layouts per ISA 7.12.2 (A: 2x16B chunks/lane; B: 2x16B chunks/lane, same k order as A).
// ---------------------------------------------------------------------------
__global__ void wmma_gemm_bf16(const bf16_t* __restrict__ A, long long bsA,
                               const bf16_t* __restrict__ W, const float* __restrict__ bias,
                               float* __restrict__ outF, bf16_t* __restrict__ outB, long long bsOut,
                               int N, int K, int relu) {
  const int lane = threadIdx.x;
  const int n0 = blockIdx.x * 16, m0 = blockIdx.y * 16, b = blockIdx.z;
  const int l15 = lane & 15;
  const bf16_t* arow = A + (size_t)b * bsA + (size_t)(m0 + l15) * K;
  const bf16_t* brow = W + (size_t)(n0 + l15) * K;
  const int klo = (lane < 16) ? 0 : 8;    // A: lanes16-31 hold K=8..15 / 24..31
  const int kbo = (lane < 16) ? 0 : 8;    // B: lanes16-31 hold K=8..15 / 24..31
  v8f acc = {};
  for (int k = 0; k < K; k += 32) {
    v8bf a0 = *(const v8bf*)(arow + k + klo);
    v8bf a1 = *(const v8bf*)(arow + k + 16 + klo);
    v8bf b0 = *(const v8bf*)(brow + k + kbo);
    v8bf b1 = *(const v8bf*)(brow + k + 16 + kbo);
    v16bf av, bv;
#pragma unroll
    for (int i = 0; i < 8; ++i) { av[i] = a0[i]; av[i+8] = a1[i]; bv[i] = b0[i]; bv[i+8] = b1[i]; }
    acc = __builtin_amdgcn_wmma_f32_16x16x32_bf16(false, av, false, bv, (short)0, acc, false, false);
  }
  const int col  = n0 + l15;
  const int row0 = m0 + ((lane >> 4) << 3);   // C/D: VGPR r -> M = r + 8*(lane>=16)
  const float bvs = bias[col];
#pragma unroll
  for (int r = 0; r < 8; ++r) {
    float v = acc[r] + bvs;
    if (relu) v = fmaxf(v, 0.0f);
    size_t off = (size_t)b * bsOut + (size_t)(row0 + r) * N + col;
    if (outF) outF[off] = v; else outB[off] = f2bf(v);
  }
}

// second tiny layer: sigmoid(H1 @ W2^T + b2), mean over S, * safety
__global__ void probs_kernel(const float* __restrict__ H1, long long bsH,
                             const float* __restrict__ W2, const float* __restrict__ b2,
                             const float* __restrict__ safety, int safStride,
                             float* __restrict__ finals, int C, int S) {
  __shared__ float sp[32];
  const int c = blockIdx.x, b = blockIdx.y, t = threadIdx.x;
  float val = 0.0f;
  if (t < S) {
    const float* h = H1 + (size_t)b * bsH + (size_t)c * C;
    const float* w = W2 + (size_t)t * C;
    float acc = b2[t];
    for (int k = 0; k < C; ++k) acc += h[k] * w[k];
    val = 1.0f / (1.0f + expf(-acc));
  }
  sp[t] = val; __syncthreads();
  if (t == 0) {
    float s = 0.f;
    for (int i = 0; i < S; ++i) s += sp[i];
    finals[(size_t)b * C + c] = (s / (float)S) * safety[b * safStride + c];
  }
}

// serial top-k over 448 candidates (LDS), emits 40 global point indices per batch
__global__ void topk_kernel(const float* __restrict__ fg, const float* __restrict__ fc,
                            const float* __restrict__ fd, const int* __restrict__ fps_idx,
                            int* __restrict__ topidx) {
  __shared__ float v[448];
  const int b = blockIdx.x, t = threadIdx.x;
  for (int i = t; i < 256; i += 32) v[i]       = fg[b*256 + i];
  for (int i = t; i < 128; i += 32) v[256 + i] = fc[b*128 + i];
  for (int i = t; i < 64;  i += 32) v[384 + i] = fd[b*64  + i];
  __syncthreads();
  if (t == 0) {
    const int Cs[3] = {256, 128, 64}, Ss[3] = {16, 16, 8}, off[3] = {0, 256, 384}, oo[3] = {0, 16, 32};
    for (int sc = 0; sc < 3; ++sc) {
      float* vv = &v[off[sc]];
      for (int j = 0; j < Ss[sc]; ++j) {
        float best = -3.4028235e38f; int bi = 0;
        for (int i = 0; i < Cs[sc]; ++i) if (vv[i] > best) { best = vv[i]; bi = i; }
        vv[bi] = -3.4028235e38f;
        topidx[b*40 + oo[sc] + j] = fps_idx[b*256 + bi];
      }
    }
  }
}

// gather 40 token rows (pad to 48 with zeros for 16-row WMMA tiles)
__global__ void gather_toks_kernel(const float* __restrict__ feat, const int* __restrict__ topidx,
                                   bf16_t* __restrict__ toks) {
  const int r = blockIdx.x, b = blockIdx.y, t = threadIdx.x;
  bf16_t* dst = toks + ((size_t)b*48 + r) * DFEAT;
  if (r < 40) {
    const int pid = topidx[b*40 + r];
    const float* src = feat + ((size_t)b*NPTS + pid) * DFEAT;
    for (int d = t; d < DFEAT; d += 256) dst[d] = f2bf(src[d]);
  } else {
    for (int d = t; d < DFEAT; d += 256) dst[d] = f2bf(0.0f);
  }
}

__global__ void ln_kernel(const float* __restrict__ h2, const float* __restrict__ g,
                          const float* __restrict__ be, float* __restrict__ out) {
  __shared__ float ss[256], sq[256];
  const int r = blockIdx.x, b = blockIdx.y, t = threadIdx.x;
  const float* h = h2 + ((size_t)b*48 + r) * DFEAT;
  float s = 0.f, q = 0.f;
  for (int d = t; d < DFEAT; d += 256) { float x = h[d]; s += x; q += x*x; }
  ss[t] = s; sq[t] = q; __syncthreads();
  for (int k = 128; k > 0; k >>= 1) { if (t < k) { ss[t] += ss[t+k]; sq[t] += sq[t+k]; } __syncthreads(); }
  const float mean = ss[0] / (float)DFEAT;
  const float var  = sq[0] / (float)DFEAT - mean * mean;
  const float inv  = rsqrtf(var + 1e-5f);
  float* o = out + ((size_t)b*40 + r) * DFEAT;
  for (int d = t; d < DFEAT; d += 256) o[d] = (h[d] - mean) * inv * g[d] + be[d];
}

extern "C" void kernel_launch(void* const* d_in, const int* in_sizes, int n_in,
                              void* d_out, int out_size, void* d_ws, size_t ws_size,
                              hipStream_t stream) {
  (void)in_sizes; (void)n_in; (void)out_size; (void)ws_size;
  const float* feat   = (const float*)d_in[0];
  const float* coords = (const float*)d_in[1];
  const float* gW1 = (const float*)d_in[2];  const float* gb1 = (const float*)d_in[3];
  const float* gW2 = (const float*)d_in[4];  const float* gb2 = (const float*)d_in[5];
  const float* cW1 = (const float*)d_in[6];  const float* cb1 = (const float*)d_in[7];
  const float* cW2 = (const float*)d_in[8];  const float* cb2 = (const float*)d_in[9];
  const float* dW1 = (const float*)d_in[10]; const float* db1 = (const float*)d_in[11];
  const float* dW2 = (const float*)d_in[12]; const float* db2 = (const float*)d_in[13];
  const float* pW1 = (const float*)d_in[14]; const float* pb1 = (const float*)d_in[15];
  const float* pW2 = (const float*)d_in[16]; const float* pb2 = (const float*)d_in[17];
  const float* lng = (const float*)d_in[18]; const float* lnb = (const float*)d_in[19];
  float* out = (float*)d_out;

  char* p = (char*)d_ws;
  auto alloc = [&](size_t bytes) -> void* {
    void* r = (void*)p; p += (bytes + 255) & ~(size_t)255; return r;
  };
  int*    fps_idx = (int*)alloc((size_t)NB*256*sizeof(int));
  float*  zstat   = (float*)alloc((size_t)NB*2*sizeof(float));
  float*  density = (float*)alloc((size_t)NB*64*sizeof(float));
  float*  sg = (float*)alloc((size_t)NB*256*4);
  float*  sc = (float*)alloc((size_t)NB*256*4);
  float*  sd = (float*)alloc((size_t)NB*64*4);
  float*  fg = (float*)alloc((size_t)NB*256*4);
  float*  fc = (float*)alloc((size_t)NB*128*4);
  float*  fd = (float*)alloc((size_t)NB*64*4);
  int*    topidx = (int*)alloc((size_t)NB*40*4);
  bf16_t* cfeat = (bf16_t*)alloc((size_t)NB*256*DFEAT*2);
  bf16_t* gW1b  = (bf16_t*)alloc((size_t)256*DFEAT*2);
  bf16_t* cW1b  = (bf16_t*)alloc((size_t)128*DFEAT*2);
  bf16_t* dW1b  = (bf16_t*)alloc((size_t)64*DFEAT*2);
  bf16_t* pW1b  = (bf16_t*)alloc((size_t)384*DFEAT*2);
  bf16_t* pW2b  = (bf16_t*)alloc((size_t)768*384*2);
  float*  H1g   = (float*)alloc((size_t)NB*256*256*4);
  float*  H1c   = (float*)alloc((size_t)NB*128*128*4);
  float*  H1d   = (float*)alloc((size_t)NB*64*64*4);
  bf16_t* toks  = (bf16_t*)alloc((size_t)NB*48*DFEAT*2);
  bf16_t* h1    = (bf16_t*)alloc((size_t)NB*48*384*2);
  float*  h2    = (float*)alloc((size_t)NB*48*DFEAT*4);

  fps_kernel<<<NB, 256, 0, stream>>>(coords, fps_idx);
  zstats_kernel<<<NB, 256, 0, stream>>>(coords, zstat);
  density_kernel<<<dim3(64, NB), 256, 0, stream>>>(coords, fps_idx, density);
  safety_kernel<<<NB, 256, 0, stream>>>(coords, fps_idx, zstat, density, sg, sc, sd);

  cvt_f32_bf16<<<(256*DFEAT + 255)/256, 256, 0, stream>>>(gW1, gW1b, 256*DFEAT);
  cvt_f32_bf16<<<(128*DFEAT + 255)/256, 256, 0, stream>>>(cW1, cW1b, 128*DFEAT);
  cvt_f32_bf16<<<( 64*DFEAT + 255)/256, 256, 0, stream>>>(dW1, dW1b, 64*DFEAT);
  cvt_f32_bf16<<<(384*DFEAT + 255)/256, 256, 0, stream>>>(pW1, pW1b, 384*DFEAT);
  cvt_f32_bf16<<<(768*384  + 255)/256, 256, 0, stream>>>(pW2, pW2b, 768*384);

  gather_cfeat_kernel<<<dim3(256, NB), 256, 0, stream>>>(feat, fps_idx, cfeat);

  // center scoring: H1 = relu(cfeat @ W1^T + b1)   (WMMA bf16 -> f32)
  wmma_gemm_bf16<<<dim3(16, 16, NB), 32, 0, stream>>>(cfeat, (long long)256*DFEAT, gW1b, gb1,
                                                      H1g, nullptr, (long long)256*256, 256, DFEAT, 1);
  wmma_gemm_bf16<<<dim3(8, 8, NB), 32, 0, stream>>>(cfeat, (long long)256*DFEAT, cW1b, cb1,
                                                    H1c, nullptr, (long long)128*128, 128, DFEAT, 1);
  wmma_gemm_bf16<<<dim3(4, 4, NB), 32, 0, stream>>>(cfeat, (long long)256*DFEAT, dW1b, db1,
                                                    H1d, nullptr, (long long)64*64, 64, DFEAT, 1);

  probs_kernel<<<dim3(256, NB), 32, 0, stream>>>(H1g, (long long)256*256, gW2, gb2, sg, 256, fg, 256, 16);
  probs_kernel<<<dim3(128, NB), 32, 0, stream>>>(H1c, (long long)128*128, cW2, cb2, sc, 256, fc, 128, 16);
  probs_kernel<<<dim3(64,  NB), 32, 0, stream>>>(H1d, (long long)64*64,   dW2, db2, sd, 64,  fd, 64,  8);

  topk_kernel<<<NB, 32, 0, stream>>>(fg, fc, fd, fps_idx, topidx);
  gather_toks_kernel<<<dim3(48, NB), 256, 0, stream>>>(feat, topidx, toks);

  // token MLP: h1 = relu(toks @ pW1^T + b1) [bf16 out], h2 = h1 @ pW2^T + b2 [f32]
  wmma_gemm_bf16<<<dim3(24, 3, NB), 32, 0, stream>>>(toks, (long long)48*DFEAT, pW1b, pb1,
                                                     nullptr, h1, (long long)48*384, 384, DFEAT, 1);
  wmma_gemm_bf16<<<dim3(48, 3, NB), 32, 0, stream>>>(h1, (long long)48*384, pW2b, pb2,
                                                     h2, nullptr, (long long)48*DFEAT, DFEAT, 384, 0);

  ln_kernel<<<dim3(40, NB), 256, 0, stream>>>(h2, lng, lnb, out);
}
